// DxPrivacy_49443663511898
// MI455X (gfx1250) — matrix operand order repaired and checked by reading, not compiled, not run on hardware
//
#include <hip/hip_runtime.h>
#include <hip/hip_bf16.h>

typedef __attribute__((ext_vector_type(16))) __bf16 v16bf;
typedef __attribute__((ext_vector_type(8)))  float  v8f;

#define M_TOT 8192      // B*S = 4*2048
#define D_DIM 4096
#define V_DIM 32000
#define BM 128
#define BN 128
#define BK 32
#define NK (D_DIM / BK)          // 128 K-steps
#define LDS_ROW_B 80             // 64 data bytes + 16 pad -> conflict-free b128 reads
#define A_TILE_B (BM * LDS_ROW_B)        // 10240 B
#define B_TILE_B (BN * LDS_ROW_B)        // 10240 B
#define BUF_B    (A_TILE_B + B_TILE_B)   // 20480 B per stage

// ---------------- init: zero per-row (key<<32|idx) argmax accumulators -----
__global__ void dx_init_best(unsigned long long* __restrict__ best) {
    int i = blockIdx.x * blockDim.x + threadIdx.x;
    if (i < M_TOT) best[i] = 0ull;   // key 0 < any real score key
}

// ---------------- prep: x_bf16 = bf16(inputs + noise), 4 elems/thread ------
__global__ void dx_prep_x(const float* __restrict__ x, const float* __restrict__ nz,
                          unsigned short* __restrict__ o) {
    size_t i = (size_t)blockIdx.x * blockDim.x + threadIdx.x;   // float4 chunk id
    const float4 a = reinterpret_cast<const float4*>(x)[i];
    const float4 b = reinterpret_cast<const float4*>(nz)[i];
    union { __bf16 h[4]; uint2 u; } p;
    p.h[0] = (__bf16)(a.x + b.x);
    p.h[1] = (__bf16)(a.y + b.y);
    p.h[2] = (__bf16)(a.z + b.z);
    p.h[3] = (__bf16)(a.w + b.w);
    reinterpret_cast<uint2*>(o)[i] = p.u;
}

__device__ inline unsigned dx_f2key(float f) {
    unsigned u = __float_as_uint(f);
    return (u & 0x80000000u) ? ~u : (u | 0x80000000u);  // monotone total order
}

// ---------------- GEMM (bf16 WMMA) + fused row-argmax ----------------------
__global__ __launch_bounds__(256)
void dx_gemm_argmax(const unsigned short* __restrict__ xbf,
                    const float* __restrict__ table,
                    unsigned long long* __restrict__ best) {
    // Single static LDS block, manually partitioned: stage s at s*BUF_B,
    // A tile at +0, B tile at +A_TILE_B. Static LDS starts at wave LDS offset 0,
    // so these byte offsets are valid VDST operands for async loads.
    __shared__ char lds_raw[2 * BUF_B];

    const int m0   = blockIdx.x * BM;          // M fastest -> N-block shared in L2
    const int n0   = blockIdx.y * BN;
    const int tid  = threadIdx.x;
    const int lane = tid & 31;
    const int wave = tid >> 5;
    const int waveM = wave & 3;                // 4 waves over M (32 rows each)
    const int waveN = wave >> 2;               // 2 waves over N (64 cols each)
    const int lcol  = lane & 15;
    const int hi16  = (lane & 16) ? 1 : 0;

    const unsigned long long xbase = (unsigned long long)(uintptr_t)xbf;

    v8f acc[2][4] = {};
    float4 breg[4];

    // ---- async A tile copy (bf16, pure move: global -> LDS, no VGPR data) ----
    auto issue_async_A = [&](int kb, int buf) {
        #pragma unroll
        for (int c = 0; c < 2; ++c) {
            int chunk = tid + c * 256;                    // 512 x 16B chunks
            int row = chunk >> 2, col = chunk & 3;
            unsigned voff = (unsigned)(((size_t)(m0 + row) * D_DIM + kb) * 2 + col * 16);
            unsigned loff = (unsigned)(buf * BUF_B + row * LDS_ROW_B + col * 16);
            asm volatile("global_load_async_to_lds_b128 %0, %1, %2"
                         :: "v"(loff), "v"(voff), "s"(xbase) : "memory");
        }
    };
    // ---- B tile: global fp32 -> regs (conversion happens at store time) ----
    auto load_B = [&](int kb) {
        #pragma unroll
        for (int c = 0; c < 4; ++c) {
            int chunk = tid + c * 256;                    // 1024 x 4-float chunks
            int row = chunk >> 3, col = chunk & 7;
            breg[c] = *reinterpret_cast<const float4*>(
                table + (size_t)(n0 + row) * D_DIM + kb + col * 4);
        }
    };
    auto store_B = [&](int buf) {
        #pragma unroll
        for (int c = 0; c < 4; ++c) {
            int chunk = tid + c * 256;
            int row = chunk >> 3, col = chunk & 7;
            union { __bf16 h[4]; uint2 u; } p;
            p.h[0] = (__bf16)breg[c].x; p.h[1] = (__bf16)breg[c].y;
            p.h[2] = (__bf16)breg[c].z; p.h[3] = (__bf16)breg[c].w;
            *reinterpret_cast<uint2*>(
                lds_raw + buf * BUF_B + A_TILE_B + row * LDS_ROW_B + col * 8) = p.u;
        }
    };

    // ---- prologue: stage tile 0 into buffer 0 ----
    issue_async_A(0, 0);
    load_B(0);
    store_B(0);
    asm volatile("s_wait_asynccnt 0" ::: "memory");
    __syncthreads();

    int cur = 0;
    for (int kb = 0; kb < NK; ++kb) {
        const int nxt = cur ^ 1;
        const bool more = (kb + 1) < NK;
        if (more) {
            issue_async_A((kb + 1) * BK, nxt);   // ASYNCcnt path, no VGPRs
            load_B((kb + 1) * BK);               // overlaps with WMMA below
        }

        // ---- fragments from LDS[cur] ----
        union Frag { v16bf v; uint4 u[2]; };
        Frag a[2], b[4];
        // A layout (16-bit 16x32): lanes 0-15 K{0..7,16..23}, lanes 16-31 K{8..15,24..31}
        #pragma unroll
        for (int i = 0; i < 2; ++i) {
            const char* base = lds_raw + cur * BUF_B
                + (waveM * 32 + i * 16 + lcol) * LDS_ROW_B + hi16 * 16;
            a[i].u[0] = *reinterpret_cast<const uint4*>(base);
            a[i].u[1] = *reinterpret_cast<const uint4*>(base + 32);
        }
        // B layout (16-bit 32x16): lanes 0-15 K{0..7,16..23}, lanes 16-31 K{8..15,24..31}
        #pragma unroll
        for (int j = 0; j < 4; ++j) {
            const char* base = lds_raw + cur * BUF_B + A_TILE_B
                + (waveN * 64 + j * 16 + lcol) * LDS_ROW_B + hi16 * 16;
            b[j].u[0] = *reinterpret_cast<const uint4*>(base);
            b[j].u[1] = *reinterpret_cast<const uint4*>(base + 32);
        }
        #pragma unroll
        for (int i = 0; i < 2; ++i)
            #pragma unroll
            for (int j = 0; j < 4; ++j)
                acc[i][j] = __builtin_amdgcn_wmma_f32_16x16x32_bf16(
                    false, a[i].v, false, b[j].v, (short)0, acc[i][j], false, false);

        if (more) store_B(nxt);                  // waits on its own global loads
        asm volatile("s_wait_asynccnt 0" ::: "memory");  // A tile landed in LDS
        __syncthreads();                          // publish nxt, retire cur reads
        cur = nxt;
    }

    // C/D layout: VGPR e holds M = e + 8*hi16, N = lane&15. Reduce this wave's
    // 64 N columns, then packed atomicMax into per-row slot.
    #pragma unroll
    for (int i = 0; i < 2; ++i) {
        #pragma unroll
        for (int e = 0; e < 8; ++e) {
            float    bv = acc[i][0][e];
            unsigned bn = (unsigned)(n0 + waveN * 64 + lcol);
            #pragma unroll
            for (int j = 1; j < 4; ++j) {
                float v = acc[i][j][e];
                unsigned n = (unsigned)(n0 + waveN * 64 + j * 16 + lcol);
                if (v > bv) { bv = v; bn = n; }
            }
            #pragma unroll
            for (int mask = 1; mask < 16; mask <<= 1) {
                float    ov = __shfl_xor(bv, mask, 32);
                unsigned on = __shfl_xor(bn, mask, 32);
                if (ov > bv || (ov == bv && on < bn)) { bv = ov; bn = on; }
            }
            if (lcol == 0) {
                int row = m0 + waveM * 32 + i * 16 + e + hi16 * 8;
                unsigned long long packed =
                    ((unsigned long long)dx_f2key(bv) << 32) | (unsigned long long)bn;
                atomicMax(&best[row], packed);
            }
        }
    }
}

// ---------------- gather: out[m, :] = table[idx[m], :] ---------------------
__global__ void dx_gather(const unsigned long long* __restrict__ best,
                          const float* __restrict__ table,
                          float* __restrict__ out) {
    int row = blockIdx.x;
    unsigned idx = (unsigned)(best[row] & 0xFFFFFFFFull);
    const float4* src = reinterpret_cast<const float4*>(table + (size_t)idx * D_DIM);
    float4*       dst = reinterpret_cast<float4*>(out + (size_t)row * D_DIM);
    for (int t = threadIdx.x; t < D_DIM / 4; t += blockDim.x)
        dst[t] = src[t];
}

extern "C" void kernel_launch(void* const* d_in, const int* in_sizes, int n_in,
                              void* d_out, int out_size, void* d_ws, size_t ws_size,
                              hipStream_t stream) {
    const float* inputs = (const float*)d_in[0];   // [4,2048,4096] f32
    const float* table  = (const float*)d_in[1];   // [32000,4096]  f32
    const float* noise  = (const float*)d_in[2];   // [4,2048,4096] f32
    float* out = (float*)d_out;                    // [4,2048,4096] f32

    // ws layout: [0, 64KB) best u64[8192]; [64KB, ..) x_bf16 (64 MB)
    unsigned long long* best = (unsigned long long*)d_ws;
    unsigned short*     xbf  = (unsigned short*)((char*)d_ws + (size_t)M_TOT * 8);

    dx_init_best<<<M_TOT / 256, 256, 0, stream>>>(best);

    // 33.55M elems, 4 per thread
    dx_prep_x<<<(M_TOT * (D_DIM / 4)) / 256, 256, 0, stream>>>(inputs, noise, xbf);

    dim3 grid(M_TOT / BM, V_DIM / BN);             // 64 x 250
    dx_gemm_argmax<<<grid, 256, 0, stream>>>(xbf, table, best);

    dx_gather<<<M_TOT, 256, 0, stream>>>(best, table, out);
}
